// SelfAttention_76802605187777
// MI455X (gfx1250) — matrix operand order repaired and checked by reading, not compiled, not run on hardware
//
#include <hip/hip_runtime.h>

// ---------------- problem constants ----------------
static constexpr int BB   = 2;
static constexpr int SS   = 2048;
static constexpr int DD   = 1024;
static constexpr int HH   = 16;
static constexpr int DQQ  = 64;          // DD / HH
static constexpr int ROWS = BB * SS;     // 4096 tokens

typedef __bf16 bf16;
typedef __attribute__((ext_vector_type(16))) __bf16 v16bf;
typedef __attribute__((ext_vector_type(8)))  __bf16 v8bf;
typedef __attribute__((ext_vector_type(8)))  float  v8f;

union Frag16 { v16bf v; v8bf h[2]; };

// float -> bf16, round-to-nearest-even (manual bit twiddling)
__device__ __forceinline__ bf16 f2bf(float f) {
  unsigned u = __builtin_bit_cast(unsigned, f);
  u += 0x7FFFu + ((u >> 16) & 1u);
  unsigned short hs = (unsigned short)(u >> 16);
  return __builtin_bit_cast(bf16, hs);
}

__device__ __forceinline__ v8f wmma_bf16(const Frag16& a, const Frag16& b, v8f c) {
  // D = A(16x32 bf16) * B(32x16 bf16) + C(16x16 f32)
  return __builtin_amdgcn_wmma_f32_16x16x32_bf16(false, a.v, false, b.v,
                                                 (short)0, c, false, false);
}

// reductions within a 16-lane half of the wave32 (xor masks 1..8 stay in-half)
__device__ __forceinline__ float hmax16(float v) {
#pragma unroll
  for (int m = 8; m; m >>= 1) v = fmaxf(v, __shfl_xor(v, m, 32));
  return v;
}
__device__ __forceinline__ float hsum16(float v) {
#pragma unroll
  for (int m = 8; m; m >>= 1) v += __shfl_xor(v, m, 32);
  return v;
}

// ---------------- kernel 1: LayerNorm (fp32 in, bf16 out) ----------------
__global__ __launch_bounds__(256)
void ln_kernel(const float* __restrict__ x, const float* __restrict__ g,
               const float* __restrict__ b, bf16* __restrict__ xn) {
  __shared__ float s_sum[8], s_sq[8];
  const int row = blockIdx.x;
  const int t = threadIdx.x;
  const float* xr = x + (size_t)row * DD;
  float4 v = *(const float4*)(xr + t * 4);
  float sum = v.x + v.y + v.z + v.w;
  float sq  = v.x * v.x + v.y * v.y + v.z * v.z + v.w * v.w;
#pragma unroll
  for (int m = 16; m; m >>= 1) {
    sum += __shfl_xor(sum, m, 32);
    sq  += __shfl_xor(sq,  m, 32);
  }
  if ((t & 31) == 0) { s_sum[t >> 5] = sum; s_sq[t >> 5] = sq; }
  __syncthreads();
  sum = 0.f; sq = 0.f;
#pragma unroll
  for (int i = 0; i < 8; ++i) { sum += s_sum[i]; sq += s_sq[i]; }
  const float mu   = sum * (1.0f / DD);
  const float var  = sq * (1.0f / DD) - mu * mu;
  const float rstd = rsqrtf(var + 1e-5f);

  bf16* out = xn + (size_t)row * DD + t * 4;
  const float* gg = g + t * 4;
  const float* bb = b + t * 4;
  float e[4] = {v.x, v.y, v.z, v.w};
#pragma unroll
  for (int i = 0; i < 4; ++i)
    out[i] = f2bf((e[i] - mu) * rstd * gg[i] + bb[i]);
}

// ---------------- kernel 2: weight transpose + convert  Wt[n][k] = W[k][n] ----------------
__global__ __launch_bounds__(256)
void wtrans_kernel(const float* __restrict__ W, bf16* __restrict__ Wt) {
  __shared__ float tile[32][33];
  const int k0 = blockIdx.x << 5, n0 = blockIdx.y << 5;
  const int tx = threadIdx.x & 31, ty = threadIdx.x >> 5;
  for (int i = ty; i < 32; i += 8)
    tile[i][tx] = W[(size_t)(k0 + i) * DD + n0 + tx];
  __syncthreads();
  for (int i = ty; i < 32; i += 8)
    Wt[(size_t)(n0 + i) * DD + k0 + tx] = f2bf(tile[tx][i]);
}

// ---------------- kernel 3/5: WMMA GEMM, one wave -> 32x64 output tile ----------------
// A: [4096 x 1024] bf16 row-major; Wt: [1024 x 1024] bf16 (transposed weight).
// Per k-step: 2 A-fragments + 4 B-fragments feed 8 WMMAs (B reused across row tiles).
// MODE 0: Q  -> [B,H,S,DQ] bf16, scaled by 1/sqrt(DQ)
// MODE 1: K  -> [B,H,S,DQ] bf16
// MODE 2: V  -> [B,H,DQ,S] bf16 (transposed for PV B-fragments)
// MODE 3: out-> fp32 [B,S,D] with bias + residual
template <int MODE>
__global__ __launch_bounds__(128)
void gemm32x64(const bf16* __restrict__ A, const bf16* __restrict__ Wt,
               const float* __restrict__ bias, const float* __restrict__ resid,
               void* __restrict__ outp) {
  const int wid   = (blockIdx.x << 2) + (threadIdx.x >> 5);
  const int m0    = (wid >> 4) << 5;      // 128 row tiles of 32
  const int n0    = (wid & 15) << 6;      // 16 col groups of 64
  const int lane  = threadIdx.x & 31;
  const int l     = lane & 15;
  const int hi    = lane >> 4;

  const bf16* aptr0 = A + (size_t)(m0 + l) * DD;
  const bf16* aptr1 = A + (size_t)(m0 + 16 + l) * DD;
  const bf16* wp[4];
#pragma unroll
  for (int t = 0; t < 4; ++t)
    wp[t] = Wt + (size_t)(n0 + t * 16 + l) * DD + hi * 8;

  v8f acc[2][4] = {};
  for (int k0 = 0; k0 < DD; k0 += 32) {
    __builtin_prefetch(aptr0 + k0 + 256, 0, 1);      // global_prefetch_b8
    Frag16 a0, a1;
    a0.h[0] = *(const v8bf*)(aptr0 + k0 + hi * 8);
    a0.h[1] = *(const v8bf*)(aptr0 + k0 + 16 + hi * 8);
    a1.h[0] = *(const v8bf*)(aptr1 + k0 + hi * 8);
    a1.h[1] = *(const v8bf*)(aptr1 + k0 + 16 + hi * 8);
#pragma unroll
    for (int t = 0; t < 4; ++t) {
      Frag16 w;
      w.h[0] = *(const v8bf*)(wp[t] + k0);
      w.h[1] = *(const v8bf*)(wp[t] + k0 + 16);
      acc[0][t] = wmma_bf16(a0, w, acc[0][t]);
      acc[1][t] = wmma_bf16(a1, w, acc[1][t]);
    }
  }

#pragma unroll
  for (int mt = 0; mt < 2; ++mt) {
#pragma unroll
    for (int r = 0; r < 8; ++r) {
      const int row = m0 + mt * 16 + r + 8 * hi;  // C-layout rows
#pragma unroll
      for (int t = 0; t < 4; ++t) {
        const int col = n0 + t * 16 + l;
        float v = acc[mt][t][r] + bias[col];
        if constexpr (MODE == 3) {
          const size_t idx = (size_t)row * DD + col;
          ((float*)outp)[idx] = v + resid[idx];
        } else {
          const int b = row >> 11, s = row & (SS - 1);
          const int h = col >> 6,  dq = col & 63;
          bf16* o = (bf16*)outp;
          if constexpr (MODE == 0)
            o[(((size_t)b * HH + h) * SS + s) * DQQ + dq] = f2bf(v * 0.125f);
          else if constexpr (MODE == 1)
            o[(((size_t)b * HH + h) * SS + s) * DQQ + dq] = f2bf(v);
          else
            o[(((size_t)b * HH + h) * DQQ + dq) * SS + s] = f2bf(v);
        }
      }
    }
  }
}

// ---------------- kernel 4: causal flash attention, one wave -> 32 queries ----------------
// Two 16-query tiles share every K and V fragment (2x compute density).
// Q pre-scaled [B,H,S,DQ]; K [B,H,S,DQ]; Vt [B,H,DQ,S]; output merged [B,S,D] bf16.
__global__ __launch_bounds__(32)
void attn_kernel(const bf16* __restrict__ Q, const bf16* __restrict__ K,
                 const bf16* __restrict__ Vt, bf16* __restrict__ M) {
  __shared__ __align__(16) bf16 pbuf[32 * 32];   // P tiles staged C-layout -> A-layout
  const int bh = blockIdx.x >> 6;                // 64 query tiles (of 32) per (b,h)
  const int q0 = (blockIdx.x & 63) << 5;
  const int lane = threadIdx.x;
  const int l = lane & 15, hi = lane >> 4;

  // Q fragments: [query tile][K-dim chunk (dq 0..31 / 32..63)]
  Frag16 qf[2][2];
#pragma unroll
  for (int qt = 0; qt < 2; ++qt) {
    const bf16* qrow = Q + ((size_t)bh * SS + q0 + qt * 16 + l) * DQQ;
    qf[qt][0].h[0] = *(const v8bf*)(qrow + hi * 8);
    qf[qt][0].h[1] = *(const v8bf*)(qrow + 16 + hi * 8);
    qf[qt][1].h[0] = *(const v8bf*)(qrow + 32 + hi * 8);
    qf[qt][1].h[1] = *(const v8bf*)(qrow + 48 + hi * 8);
  }

  v8f o[2][4] = {};
  float mrow[2][8], lrow[2][8];
#pragma unroll
  for (int qt = 0; qt < 2; ++qt)
#pragma unroll
    for (int r = 0; r < 8; ++r) { mrow[qt][r] = -3.0e38f; lrow[qt][r] = 0.f; }

  const int nkb = (q0 >> 5) + 1;                 // 32-key blocks covering keys <= q0+31
  for (int j = 0; j < nkb; ++j) {
    const int kb = j << 5;

    // scores: s[qt][kt], K-depth 64 = 2 WMMAs each; K frags shared across query tiles
    v8f s[2][2] = {};
#pragma unroll
    for (int kk = 0; kk < 2; ++kk) {
      const bf16* kr0 = K + ((size_t)bh * SS + kb + l) * DQQ + kk * 32 + hi * 8;
      const bf16* kr1 = kr0 + 16 * DQQ;
      Frag16 kf0, kf1;
      kf0.h[0] = *(const v8bf*)kr0; kf0.h[1] = *(const v8bf*)(kr0 + 16);
      kf1.h[0] = *(const v8bf*)kr1; kf1.h[1] = *(const v8bf*)(kr1 + 16);
      s[0][0] = wmma_bf16(qf[0][kk], kf0, s[0][0]);
      s[0][1] = wmma_bf16(qf[0][kk], kf1, s[0][1]);
      s[1][0] = wmma_bf16(qf[1][kk], kf0, s[1][0]);
      s[1][1] = wmma_bf16(qf[1][kk], kf1, s[1][1]);
    }

    // causal mask + online softmax per query tile (rows replicated across 16-lane half)
    float fac[2][8];
#pragma unroll
    for (int qt = 0; qt < 2; ++qt) {
#pragma unroll
      for (int r = 0; r < 8; ++r) {
        const int qi = q0 + qt * 16 + r + 8 * hi;
        if (kb + l      > qi) s[qt][0][r] = -3.0e38f;
        if (kb + 16 + l > qi) s[qt][1][r] = -3.0e38f;
        const float mx   = hmax16(fmaxf(s[qt][0][r], s[qt][1][r]));
        const float mnew = fmaxf(mrow[qt][r], mx);
        fac[qt][r]  = __expf(mrow[qt][r] - mnew);
        mrow[qt][r] = mnew;
        const float p0 = __expf(s[qt][0][r] - mnew);
        const float p1 = __expf(s[qt][1][r] - mnew);
        s[qt][0][r] = p0; s[qt][1][r] = p1;
        lrow[qt][r] = lrow[qt][r] * fac[qt][r] + hsum16(p0 + p1);
      }
#pragma unroll
      for (int t = 0; t < 4; ++t)
#pragma unroll
        for (int r = 0; r < 8; ++r)
          o[qt][t][r] *= fac[qt][r];

      // C-layout -> A-layout transpose of P(16x32) through LDS
#pragma unroll
      for (int r = 0; r < 8; ++r) {
        const int m = qt * 16 + r + 8 * hi;
        pbuf[m * 32 + l]      = f2bf(s[qt][0][r]);
        pbuf[m * 32 + 16 + l] = f2bf(s[qt][1][r]);
      }
    }
    asm volatile("s_wait_dscnt 0x0" ::: "memory");
    __builtin_amdgcn_wave_barrier();
    Frag16 pa0, pa1;
    pa0.h[0] = *(const v8bf*)(pbuf + l * 32 + hi * 8);
    pa0.h[1] = *(const v8bf*)(pbuf + l * 32 + 16 + hi * 8);
    pa1.h[0] = *(const v8bf*)(pbuf + (16 + l) * 32 + hi * 8);
    pa1.h[1] = *(const v8bf*)(pbuf + (16 + l) * 32 + 16 + hi * 8);

    // O += P(16x32) @ V(32x64): V fragments shared across both query tiles
#pragma unroll
    for (int t = 0; t < 4; ++t) {
      const bf16* vr = Vt + ((size_t)bh * DQQ + t * 16 + l) * SS + kb + hi * 8;
      Frag16 vf;
      vf.h[0] = *(const v8bf*)vr; vf.h[1] = *(const v8bf*)(vr + 16);
      o[0][t] = wmma_bf16(pa0, vf, o[0][t]);
      o[1][t] = wmma_bf16(pa1, vf, o[1][t]);
    }
    __builtin_amdgcn_wave_barrier();   // LDS reads stay before next iteration's writes
  }

  const int b = bh >> 4, h = bh & 15;
#pragma unroll
  for (int qt = 0; qt < 2; ++qt)
#pragma unroll
    for (int r = 0; r < 8; ++r) {
      const int qi = q0 + qt * 16 + r + 8 * hi;
      const float inv = 1.0f / lrow[qt][r];
      bf16* mr = M + ((size_t)b * SS + qi) * DD + h * DQQ;
#pragma unroll
      for (int t = 0; t < 4; ++t)
        mr[t * 16 + l] = f2bf(o[qt][t][r] * inv);
    }
}

// ---------------- host orchestration ----------------
extern "C" void kernel_launch(void* const* d_in, const int* in_sizes, int n_in,
                              void* d_out, int out_size, void* d_ws, size_t ws_size,
                              hipStream_t stream) {
  (void)in_sizes; (void)n_in; (void)out_size; (void)ws_size;
  const float* x   = (const float*)d_in[0];
  const float* lng = (const float*)d_in[1];
  const float* lnb = (const float*)d_in[2];
  const float* Wq  = (const float*)d_in[3];
  const float* bq  = (const float*)d_in[4];
  const float* Wk  = (const float*)d_in[5];
  const float* bk  = (const float*)d_in[6];
  const float* Wv  = (const float*)d_in[7];
  const float* bv  = (const float*)d_in[8];
  const float* Wo  = (const float*)d_in[9];
  const float* bo  = (const float*)d_in[10];

  // workspace carve-up (bf16): xn + 4 weights + Q + K + Vt + merged ~= 48 MiB
  char* w = (char*)d_ws;
  auto take = [&](size_t elems) {
    bf16* p = (bf16*)w;
    w += ((elems * sizeof(bf16)) + 255) & ~(size_t)255;
    return p;
  };
  bf16* xn  = take((size_t)ROWS * DD);
  bf16* Wqt = take((size_t)DD * DD);
  bf16* Wkt = take((size_t)DD * DD);
  bf16* Wvt = take((size_t)DD * DD);
  bf16* Wot = take((size_t)DD * DD);
  bf16* Qb  = take((size_t)ROWS * DD);
  bf16* Kb  = take((size_t)ROWS * DD);
  bf16* Vtb = take((size_t)ROWS * DD);
  bf16* Mb  = take((size_t)ROWS * DD);

  ln_kernel<<<ROWS, 256, 0, stream>>>(x, lng, lnb, xn);

  dim3 tg(DD / 32, DD / 32);
  wtrans_kernel<<<tg, 256, 0, stream>>>(Wq, Wqt);
  wtrans_kernel<<<tg, 256, 0, stream>>>(Wk, Wkt);
  wtrans_kernel<<<tg, 256, 0, stream>>>(Wv, Wvt);
  wtrans_kernel<<<tg, 256, 0, stream>>>(Wo, Wot);

  const int gblk = (ROWS / 32) * (DD / 64) / 4;   // 2048 waves / 4 waves per block
  gemm32x64<0><<<gblk, 128, 0, stream>>>(xn, Wqt, bq, nullptr, Qb);
  gemm32x64<1><<<gblk, 128, 0, stream>>>(xn, Wkt, bk, nullptr, Kb);
  gemm32x64<2><<<gblk, 128, 0, stream>>>(xn, Wvt, bv, nullptr, Vtb);

  attn_kernel<<<BB * HH * (SS / 32), 32, 0, stream>>>(Qb, Kb, Vtb, Mb);

  gemm32x64<3><<<gblk, 128, 0, stream>>>(Mb, Wot, bo, x, d_out);
}
